// TTA_9912784519320
// MI455X (gfx1250) — matrix operand order repaired and checked by reading, not compiled, not run on hardware
//
#include <hip/hip_runtime.h>
#include <math.h>

#define B_ 4096
#define L_ 4096
#define D_ 1024
#define C_ 64
#define N_ 8192
#define NB_ (N_ + B_)
#define TOPK_ 5
#define EPS_ 1e-12f
#define NEGINF_ (-3.0e38f)

typedef __attribute__((ext_vector_type(16))) _Float16 v16h;
typedef __attribute__((ext_vector_type(8)))  _Float16 v8h;
typedef __attribute__((ext_vector_type(8)))  float    v8f;

// accumulator slots in ws
#define ACC_SUM_E     0
#define ACC_SUM_PMAX  1
#define ACC_SUM_WC    2
#define ACC_TSALLIS   3
#define ACC_L2NUM     4
#define ACC_L3NUM     5
#define ACC_COUNT     8

// ---------------- WMMA fragment helpers (wave32, 16x16x32 f16) ----------------
// A-frag layout (16x32 f16): lane<16 holds row M=lane, K = {k0..k0+7, k0+16..k0+23}
//                            lane>=16 holds row M=lane-16, K = {k0+8..k0+15, k0+24..k0+31}
__device__ __forceinline__ v16h load_a_f32(const float* __restrict__ rowp, int k0, int lane) {
  int hl = (lane >> 4) & 1;
  const float* p = rowp + k0 + hl * 8;
  v16h r;
#pragma unroll
  for (int i = 0; i < 8; ++i) r[i] = (_Float16)p[i];
#pragma unroll
  for (int i = 0; i < 8; ++i) r[i + 8] = (_Float16)p[i + 16];
  return r;
}

__device__ __forceinline__ v16h load_a_f16(const _Float16* __restrict__ rowp, int k0, int lane) {
  int hl = (lane >> 4) & 1;
  const _Float16* p = rowp + k0 + hl * 8;
  v8h lo = *(const v8h*)p;
  v8h hi = *(const v8h*)(p + 16);
  v16h r;
#pragma unroll
  for (int i = 0; i < 8; ++i) { r[i] = lo[i]; r[i + 8] = hi[i]; }
  return r;
}

// A-frag sourced from an LDS panel As[row*D_ + k] (row = local row in panel)
__device__ __forceinline__ v16h load_a_lds(const _Float16* As, int row, int k0, int lane) {
  int hl = (lane >> 4) & 1;
  const _Float16* p = As + row * D_ + k0 + hl * 8;
  v8h lo = *(const v8h*)p;
  v8h hi = *(const v8h*)(p + 16);
  v16h r;
#pragma unroll
  for (int i = 0; i < 8; ++i) { r[i] = lo[i]; r[i + 8] = hi[i]; }
  return r;
}

// B-frag layout (32x16 f16): lane<16 holds col N=lane, K = {k0..k0+7, k0+16..k0+23}
//                            lane>=16 holds col N=lane-16, K = {k0+8..k0+15, k0+24..k0+31}
// Bn is the B operand stored N-major: Bn[n*ldb + k] = B[k][n]
__device__ __forceinline__ v16h load_b_f16(const _Float16* __restrict__ Bn, int ldb,
                                           int n0, int k0, int lane) {
  int n = lane & 15, hl = (lane >> 4) & 1;
  const _Float16* p = Bn + (size_t)(n0 + n) * ldb + k0 + hl * 8;
  v8h lo = *(const v8h*)p;
  v8h hi = *(const v8h*)(p + 16);
  v16h r;
#pragma unroll
  for (int i = 0; i < 8; ++i) { r[i] = lo[i]; r[i + 8] = hi[i]; }
  return r;
}

__device__ __forceinline__ v8f wmma16(v16h a, v16h b, v8f c) {
  return __builtin_amdgcn_wmma_f32_16x16x32_f16(false, a, false, b, (short)0, c, false, false);
}

// ---------------- small kernels ----------------
__global__ void k_zero(float* acc, float* pmacc) {
  if (threadIdx.x < ACC_COUNT) acc[threadIdx.x] = 0.f;
  if (threadIdx.x < C_) pmacc[threadIdx.x] = 0.f;
}

// per-row input-energy entropy: e[b] = -sum q*log(q+1e-30), q = x^2/sum(x^2)
__global__ void k_entropy(const float* __restrict__ x, float* __restrict__ e, float* acc) {
  __shared__ float red[256];
  int b = blockIdx.x;
  const float* row = x + (size_t)b * L_;
  float s = 0.f;
  for (int i = threadIdx.x; i < L_; i += 256) { float v = row[i]; s += v * v; }
  red[threadIdx.x] = s; __syncthreads();
  for (int st = 128; st > 0; st >>= 1) {
    if (threadIdx.x < st) red[threadIdx.x] += red[threadIdx.x + st];
    __syncthreads();
  }
  float S = fmaxf(red[0], EPS_);
  __syncthreads();
  float h = 0.f;
  for (int i = threadIdx.x; i < L_; i += 256) {
    float v = row[i]; float q = v * v / S; h -= q * logf(q + 1e-30f);
  }
  red[threadIdx.x] = h; __syncthreads();
  for (int st = 128; st > 0; st >>= 1) {
    if (threadIdx.x < st) red[threadIdx.x] += red[threadIdx.x + st];
    __syncthreads();
  }
  if (threadIdx.x == 0) { e[b] = red[0]; atomicAdd(&acc[ACC_SUM_E], red[0]); }
}

// WfT[d*L + l] = f16(Wf[l*D + d])  (LDS-tiled transpose)
__global__ void k_transpose_wf(const float* __restrict__ Wf, _Float16* __restrict__ WfT) {
  __shared__ float t[32][33];
  int l0 = blockIdx.x * 32, d0 = blockIdx.y * 32;
  int tx = threadIdx.x, ty = threadIdx.y;
  for (int r = ty; r < 32; r += 8)
    t[r][tx] = Wf[(size_t)(l0 + r) * D_ + d0 + tx];
  __syncthreads();
  for (int r = ty; r < 32; r += 8)
    WfT[(size_t)(d0 + r) * L_ + l0 + tx] = (_Float16)t[tx][r];
}

__global__ void k_cvt_f16(const float* __restrict__ src, _Float16* __restrict__ dst, int n) {
  int i = blockIdx.x * blockDim.x + threadIdx.x;
  if (i < n) dst[i] = (_Float16)src[i];
}

// ---------------- GEMM: wave computes 32(M) x 64(N), K-loop step 32, B frags reused 2x ----
__global__ void k_gemm_f32A(const float* __restrict__ A, int lda,
                            const _Float16* __restrict__ Bn, int ldb,
                            float* __restrict__ Cm, int ldc,
                            int Kdim, const float* __restrict__ bias) {
  int lane = threadIdx.x & 31;
  int wv = threadIdx.x >> 5;
  int m0 = (blockIdx.y * 4 + wv) * 32;
  int n0 = blockIdx.x * 64;
  const float* rowp0 = A + (size_t)(m0 + (lane & 15)) * lda;
  const float* rowp1 = rowp0 + (size_t)16 * lda;
  v8f c00 = {}, c01 = {}, c02 = {}, c03 = {};
  v8f c10 = {}, c11 = {}, c12 = {}, c13 = {};
  for (int k0 = 0; k0 < Kdim; k0 += 32) {
    v16h a0 = load_a_f32(rowp0, k0, lane);
    v16h a1 = load_a_f32(rowp1, k0, lane);
    v16h b0 = load_b_f16(Bn, ldb, n0 +  0, k0, lane);
    v16h b1 = load_b_f16(Bn, ldb, n0 + 16, k0, lane);
    v16h b2 = load_b_f16(Bn, ldb, n0 + 32, k0, lane);
    v16h b3 = load_b_f16(Bn, ldb, n0 + 48, k0, lane);
    c00 = wmma16(a0, b0, c00); c10 = wmma16(a1, b0, c10);
    c01 = wmma16(a0, b1, c01); c11 = wmma16(a1, b1, c11);
    c02 = wmma16(a0, b2, c02); c12 = wmma16(a1, b2, c12);
    c03 = wmma16(a0, b3, c03); c13 = wmma16(a1, b3, c13);
  }
  int n = lane & 15, hl = lane >> 4;
  float bb0 = bias ? bias[n0 + n]      : 0.f;
  float bb1 = bias ? bias[n0 + 16 + n] : 0.f;
  float bb2 = bias ? bias[n0 + 32 + n] : 0.f;
  float bb3 = bias ? bias[n0 + 48 + n] : 0.f;
#pragma unroll
  for (int v = 0; v < 8; ++v) {
    size_t m = (size_t)(m0 + v + 8 * hl);
    Cm[m * ldc + n0 + n]      = c00[v] + bb0;
    Cm[m * ldc + n0 + 16 + n] = c01[v] + bb1;
    Cm[m * ldc + n0 + 32 + n] = c02[v] + bb2;
    Cm[m * ldc + n0 + 48 + n] = c03[v] + bb3;
    size_t m2 = m + 16;
    Cm[m2 * ldc + n0 + n]      = c10[v] + bb0;
    Cm[m2 * ldc + n0 + 16 + n] = c11[v] + bb1;
    Cm[m2 * ldc + n0 + 32 + n] = c12[v] + bb2;
    Cm[m2 * ldc + n0 + 48 + n] = c13[v] + bb3;
  }
}

// small f16-A GEMM (wave computes 16x64), used for z2 = l2n(feats) @ protos
__global__ void k_gemm_f16A(const _Float16* __restrict__ A, int lda,
                            const _Float16* __restrict__ Bn, int ldb,
                            float* __restrict__ Cm, int ldc, int Kdim) {
  int lane = threadIdx.x & 31;
  int wv = threadIdx.x >> 5;
  int m0 = (blockIdx.y * 4 + wv) * 16;
  int n0 = blockIdx.x * 64;
  const _Float16* rowp = A + (size_t)(m0 + (lane & 15)) * lda;
  v8f c0 = {}, c1 = {}, c2 = {}, c3 = {};
  for (int k0 = 0; k0 < Kdim; k0 += 32) {
    v16h a  = load_a_f16(rowp, k0, lane);
    v16h b0 = load_b_f16(Bn, ldb, n0 +  0, k0, lane);
    v16h b1 = load_b_f16(Bn, ldb, n0 + 16, k0, lane);
    v16h b2 = load_b_f16(Bn, ldb, n0 + 32, k0, lane);
    v16h b3 = load_b_f16(Bn, ldb, n0 + 48, k0, lane);
    c0 = wmma16(a, b0, c0);
    c1 = wmma16(a, b1, c1);
    c2 = wmma16(a, b2, c2);
    c3 = wmma16(a, b3, c3);
  }
  int n = lane & 15, hl = lane >> 4;
#pragma unroll
  for (int v = 0; v < 8; ++v) {
    size_t m = (size_t)(m0 + v + 8 * hl);
    Cm[m * ldc + n0 + n]      = c0[v];
    Cm[m * ldc + n0 + 16 + n] = c1[v];
    Cm[m * ldc + n0 + 32 + n] = c2[v];
    Cm[m * ldc + n0 + 48 + n] = c3[v];
  }
}

// softmax / log-softmax / argmax over C=64 per row
__global__ void k_softmax(const float* __restrict__ logits, float* probs, float* logp,
                          int* pred, float* pmax, float* acc) {
  __shared__ float rv[64]; __shared__ int ri[64];
  int b = blockIdx.x, c = threadIdx.x;
  float v = logits[(size_t)b * C_ + c];
  rv[c] = v; ri[c] = c; __syncthreads();
  for (int st = 32; st > 0; st >>= 1) {
    if (c < st && rv[c + st] > rv[c]) { rv[c] = rv[c + st]; ri[c] = ri[c + st]; }
    __syncthreads();
  }
  float mx = rv[0]; int am = ri[0];
  __syncthreads();
  float pexp = expf(v - mx);
  rv[c] = pexp; __syncthreads();
  for (int st = 32; st > 0; st >>= 1) {
    if (c < st) rv[c] += rv[c + st];
    __syncthreads();
  }
  float s = rv[0];
  probs[(size_t)b * C_ + c] = pexp / s;
  logp[(size_t)b * C_ + c]  = v - mx - logf(s);
  if (c == 0) {
    pred[b] = am;
    float pm = 1.f / s;           // max prob
    pmax[b] = pm;
    atomicAdd(&acc[ACC_SUM_PMAX], pm);
  }
}

__global__ void k_wc(const float* __restrict__ e, const float* __restrict__ pmax,
                     float* __restrict__ w_c, float* acc) {
  __shared__ float red[256];
  int b = blockIdx.x * 256 + threadIdx.x;
  float emean = acc[ACC_SUM_E] * (1.0f / (float)B_);
  float pmean = acc[ACC_SUM_PMAX] * (1.0f / (float)B_);
  float w = ((e[b] < emean) && (pmax[b] >= pmean)) ? 1.f : 0.f;
  w_c[b] = w;
  red[threadIdx.x] = w; __syncthreads();
  for (int st = 128; st > 0; st >>= 1) {
    if (threadIdx.x < st) red[threadIdx.x] += red[threadIdx.x + st];
    __syncthreads();
  }
  if (threadIdx.x == 0) atomicAdd(&acc[ACC_SUM_WC], red[0]);
}

__global__ void k_m(const float* __restrict__ probs, const float* __restrict__ w_c,
                    const float* acc, float* __restrict__ m) {
  __shared__ float red[256];
  int c = blockIdx.x;
  float s = 0.f;
  for (int b = threadIdx.x; b < B_; b += 256)
    s += probs[(size_t)b * C_ + c] * w_c[b];
  red[threadIdx.x] = s; __syncthreads();
  for (int st = 128; st > 0; st >>= 1) {
    if (threadIdx.x < st) red[threadIdx.x] += red[threadIdx.x + st];
    __syncthreads();
  }
  if (threadIdx.x == 0) m[c] = red[0] / fmaxf(acc[ACC_SUM_WC], 1.f);
}

__global__ void k_loss1(const float* __restrict__ probs, const float* __restrict__ m,
                        float* pmacc, float* acc) {
  __shared__ float r1[64]; __shared__ float r2[64];
  int b = blockIdx.x, c = threadIdx.x;
  float raw = probs[(size_t)b * C_ + c] / fmaxf(m[c], EPS_);
  r1[c] = raw; r2[c] = raw * raw; __syncthreads();
  for (int st = 32; st > 0; st >>= 1) {
    if (c < st) { r1[c] += r1[c + st]; r2[c] += r2[c + st]; }
    __syncthreads();
  }
  float s = r1[0], ss = r2[0];
  float pb = raw / s;
  atomicAdd(&pmacc[c], pb);
  if (c == 0) atomicAdd(&acc[ACC_TSALLIS], 1.f - ss / (s * s));  // ALPHA=2
}

// normalize sup_all rows into f16; extract per-row class + weight for prototype build
__global__ void k_supnorm(const float* __restrict__ sup_old, const float* __restrict__ feats,
                          const float* __restrict__ lab_old, const int* __restrict__ pred,
                          const float* __restrict__ w_c,
                          _Float16* __restrict__ supN, int* __restrict__ ci,
                          float* __restrict__ wi) {
  __shared__ float red[256]; __shared__ float lv[64]; __shared__ int li[64];
  int i = blockIdx.x;
  const float* src = (i < N_) ? (sup_old + (size_t)i * D_) : (feats + (size_t)(i - N_) * D_);
  float s = 0.f;
  for (int d = threadIdx.x; d < D_; d += 256) { float v = src[d]; s += v * v; }
  red[threadIdx.x] = s; __syncthreads();
  for (int st = 128; st > 0; st >>= 1) {
    if (threadIdx.x < st) red[threadIdx.x] += red[threadIdx.x + st];
    __syncthreads();
  }
  float nrm = fmaxf(sqrtf(red[0]), EPS_);
  for (int d = threadIdx.x; d < D_; d += 256)
    supN[(size_t)i * D_ + d] = (_Float16)(src[d] / nrm);
  if (i < N_) {
    if (threadIdx.x < 64) { lv[threadIdx.x] = lab_old[(size_t)i * C_ + threadIdx.x]; li[threadIdx.x] = threadIdx.x; }
    __syncthreads();
    for (int st = 32; st > 0; st >>= 1) {
      if (threadIdx.x < st && lv[threadIdx.x + st] > lv[threadIdx.x]) {
        lv[threadIdx.x] = lv[threadIdx.x + st]; li[threadIdx.x] = li[threadIdx.x + st];
      }
      __syncthreads();
    }
    if (threadIdx.x == 0) { ci[i] = li[0]; wi[i] = 1.f; }
  } else {
    if (threadIdx.x == 0) { ci[i] = pred[i - N_]; wi[i] = w_c[i - N_]; }
  }
}

// pu[c][d] = sum_i (ci[i]==c) * wi[i] * supN[i][d]   (column scaling cancels in l2n)
__global__ void k_protos(const _Float16* __restrict__ supN, const int* __restrict__ ci,
                         const float* __restrict__ wi, float* __restrict__ pu) {
  int c = blockIdx.y;
  int d = blockIdx.x * 256 + threadIdx.x;
  float a = 0.f;
  for (int i = 0; i < NB_; ++i) {
    float w = wi[i];
    if (ci[i] == c && w > 0.f) a += w * (float)supN[(size_t)i * D_ + d];
  }
  pu[(size_t)c * D_ + d] = a;
}

__global__ void k_protos_norm(const float* __restrict__ pu, _Float16* __restrict__ protosT) {
  __shared__ float red[256];
  int c = blockIdx.x;
  float s = 0.f;
  for (int d = threadIdx.x; d < D_; d += 256) { float v = pu[(size_t)c * D_ + d]; s += v * v; }
  red[threadIdx.x] = s; __syncthreads();
  for (int st = 128; st > 0; st >>= 1) {
    if (threadIdx.x < st) red[threadIdx.x] += red[threadIdx.x + st];
    __syncthreads();
  }
  float nrm = fmaxf(sqrtf(red[0]), EPS_);
  for (int d = threadIdx.x; d < D_; d += 256)
    protosT[(size_t)c * D_ + d] = (_Float16)(pu[(size_t)c * D_ + d] / nrm);
}

__global__ void k_loss2(const float* __restrict__ z2, const int* __restrict__ pred,
                        const float* __restrict__ w_c, float* acc) {
  __shared__ float rv[64];
  int b = blockIdx.x, c = threadIdx.x;
  float v = z2[(size_t)b * C_ + c];
  rv[c] = v; __syncthreads();
  for (int st = 32; st > 0; st >>= 1) {
    if (c < st) rv[c] = fmaxf(rv[c], rv[c + st]);
    __syncthreads();
  }
  float mx = rv[0]; __syncthreads();
  rv[c] = expf(v - mx); __syncthreads();
  for (int st = 32; st > 0; st >>= 1) {
    if (c < st) rv[c] += rv[c + st];
    __syncthreads();
  }
  float s = rv[0];
  if (c == 0) {
    float lp = z2[(size_t)b * C_ + pred[b]] - mx - logf(s);
    atomicAdd(&acc[ACC_L2NUM], -lp * w_c[b]);
  }
}

// Fused sim GEMM (103 GFLOP) + per-row top-5. Block = 4 waves, 32 rows.
// A panel (32x1024 f16 = 64KB) staged in LDS once; each wave owns NB/4 columns,
// 2 row-subtiles share each B fragment. Per-lane top-5 (one row per lane),
// merged across waves via LDS at the end. sim never touches global memory.
#define SIM_LDS_BYTES (65536 + 8704 + 2560 + 2560)
__global__ void k_sim_topk(const _Float16* __restrict__ featsN, const _Float16* __restrict__ supN,
                           const float* __restrict__ w_c, int* __restrict__ topidx) {
  extern __shared__ char smem[];
  _Float16* As = (_Float16*)smem;                                     // 32*1024 f16
  float (*tileB)[2][16][17] = (float (*)[2][16][17])(smem + 65536);   // [4][2][16][17]
  float (*mvAll)[20] = (float (*)[20])(smem + 65536 + 8704);          // [32][20]
  int   (*miAll)[20] = (int   (*)[20])(smem + 65536 + 8704 + 2560);   // [32][20]
  int lane = threadIdx.x & 31;
  int wv = threadIdx.x >> 5;
  int m0 = blockIdx.x * 32;
  // cooperative stage of the contiguous 32-row A panel
  {
    const v8h* g = (const v8h*)(featsN + (size_t)m0 * D_);
    v8h* l = (v8h*)As;
    for (int i = threadIdx.x; i < (32 * D_) / 8; i += 128) l[i] = g[i];
  }
  __syncthreads();
  float v0 = NEGINF_, v1 = NEGINF_, v2 = NEGINF_, v3 = NEGINF_, v4 = NEGINF_;
  int   i0 = 0, i1 = 0, i2 = 0, i3 = 0, i4 = 0;
  float mnv = NEGINF_; int mnp = 0;
  int row = lane;                 // this lane's row (0..31) within the block
  int sA = row >> 4, lrA = row & 15;
  int cbeg = wv * (NB_ / 4), cend = cbeg + (NB_ / 4);
  for (int j0 = cbeg; j0 < cend; j0 += 16) {
    if (j0 + 16 < cend) {   // prefetch next B tile (global_prefetch_b8)
      const _Float16* pf = supN + (size_t)(j0 + 16 + (lane & 15)) * D_ + (lane >> 4) * 512;
      __builtin_prefetch(pf, 0, 1);
    }
    v8f a0 = {}, a1 = {};
    for (int k0 = 0; k0 < D_; k0 += 32) {
      v16h bfr = load_b_f16(supN, D_, j0, k0, lane);
      v16h af0 = load_a_lds(As, (lane & 15), k0, lane);
      v16h af1 = load_a_lds(As, 16 + (lane & 15), k0, lane);
      a0 = wmma16(af0, bfr, a0);
      a1 = wmma16(af1, bfr, a1);
    }
    int n = lane & 15, hl = lane >> 4;
#pragma unroll
    for (int v = 0; v < 8; ++v) {
      tileB[wv][0][v + 8 * hl][n] = a0[v];
      tileB[wv][1][v + 8 * hl][n] = a1[v];
    }
    __syncthreads();
#pragma unroll
    for (int cc = 0; cc < 16; ++cc) {
      int j = j0 + cc;
      int jj = (j >= N_) ? (j - N_) : 0;
      bool valid = (j < N_) || (w_c[jj] > 0.f);
      float vle = valid ? tileB[wv][sA][lrA][cc] : NEGINF_;
      if (vle > mnv) {
        if      (mnp == 0) { v0 = vle; i0 = j; }
        else if (mnp == 1) { v1 = vle; i1 = j; }
        else if (mnp == 2) { v2 = vle; i2 = j; }
        else if (mnp == 3) { v3 = vle; i3 = j; }
        else               { v4 = vle; i4 = j; }
        mnv = v0; mnp = 0;
        if (v1 < mnv) { mnv = v1; mnp = 1; }
        if (v2 < mnv) { mnv = v2; mnp = 2; }
        if (v3 < mnv) { mnv = v3; mnp = 3; }
        if (v4 < mnv) { mnv = v4; mnp = 4; }
      }
    }
    __syncthreads();
  }
  mvAll[row][wv * 5 + 0] = v0; miAll[row][wv * 5 + 0] = i0;
  mvAll[row][wv * 5 + 1] = v1; miAll[row][wv * 5 + 1] = i1;
  mvAll[row][wv * 5 + 2] = v2; miAll[row][wv * 5 + 2] = i2;
  mvAll[row][wv * 5 + 3] = v3; miAll[row][wv * 5 + 3] = i3;
  mvAll[row][wv * 5 + 4] = v4; miAll[row][wv * 5 + 4] = i4;
  __syncthreads();
  if (threadIdx.x < 32) {
    float* cv = &mvAll[threadIdx.x][0];
    int*   cx = &miAll[threadIdx.x][0];
    for (int s = 0; s < TOPK_; ++s) {
      float best = NEGINF_; int bp = 0;
      for (int t = 0; t < 20; ++t)
        if (cv[t] > best) { best = cv[t]; bp = t; }
      topidx[(size_t)(m0 + threadIdx.x) * TOPK_ + s] = cx[bp];
      cv[bp] = NEGINF_;
    }
  }
}

__global__ void k_ncl(const int* __restrict__ topidx, const float* __restrict__ sco_old,
                      const float* __restrict__ probs, const float* __restrict__ logp,
                      const float* __restrict__ w_c, float* acc) {
  __shared__ float red[64];
  int b = blockIdx.x, c = threadIdx.x;
  float t = 0.f;
#pragma unroll
  for (int k = 0; k < TOPK_; ++k) {
    int j = topidx[(size_t)b * TOPK_ + k];
    t += (j < N_) ? sco_old[(size_t)j * C_ + c] : probs[(size_t)(j - N_) * C_ + c];
  }
  t *= (1.0f / (float)TOPK_);
  red[c] = -t * logp[(size_t)b * C_ + c];
  __syncthreads();
  for (int st = 32; st > 0; st >>= 1) {
    if (c < st) red[c] += red[c + st];
    __syncthreads();
  }
  if (c == 0) atomicAdd(&acc[ACC_L3NUM], red[0] * (1.f - w_c[b]));
}

__global__ void k_final(const float* acc, const float* pmacc, float* out) {
  __shared__ float red[64];
  int c = threadIdx.x;
  float pm = pmacc[c] * (1.0f / (float)B_);
  red[c] = pm * logf(pm + EPS_);
  __syncthreads();
  for (int st = 32; st > 0; st >>= 1) {
    if (c < st) red[c] += red[c + st];
    __syncthreads();
  }
  if (c == 0) {
    float divv = -red[0];
    float ts = acc[ACC_TSALLIS] * (1.0f / (float)B_);
    float loss1 = ts - divv;
    float swc = acc[ACC_SUM_WC];
    float loss2 = acc[ACC_L2NUM] / fmaxf(swc, 1.f);
    float loss3 = acc[ACC_L3NUM] / fmaxf((float)B_ - swc, 1.f);
    out[0] = loss1 + loss2 + loss3;
  }
}

// ---------------- host ----------------
extern "C" void kernel_launch(void* const* d_in, const int* in_sizes, int n_in,
                              void* d_out, int out_size, void* d_ws, size_t ws_size,
                              hipStream_t stream) {
  (void)in_sizes; (void)n_in; (void)out_size; (void)ws_size;
  const float* x       = (const float*)d_in[0];
  // d_in[1] = y (unused by the loss)
  const float* Wf      = (const float*)d_in[2];
  const float* Wc      = (const float*)d_in[3];
  const float* bc      = (const float*)d_in[4];
  const float* sup_old = (const float*)d_in[5];
  const float* lab_old = (const float*)d_in[6];
  const float* sco_old = (const float*)d_in[7];
  float* out = (float*)d_out;

  char* p = (char*)d_ws;
  auto alloc = [&](size_t bytes) -> char* {
    char* r = p; p += (bytes + 255) & ~(size_t)255; return r;
  };
  float*    feats   = (float*)   alloc((size_t)B_ * D_ * 4);
  _Float16* WfT     = (_Float16*)alloc((size_t)D_ * L_ * 2);
  _Float16* Wch     = (_Float16*)alloc((size_t)C_ * D_ * 2);
  float*    logits  = (float*)   alloc((size_t)B_ * C_ * 4);
  float*    probs   = (float*)   alloc((size_t)B_ * C_ * 4);
  float*    logp    = (float*)   alloc((size_t)B_ * C_ * 4);
  int*      pred    = (int*)     alloc((size_t)B_ * 4);
  float*    pmaxv   = (float*)   alloc((size_t)B_ * 4);
  float*    ev      = (float*)   alloc((size_t)B_ * 4);
  float*    wcv     = (float*)   alloc((size_t)B_ * 4);
  _Float16* supN    = (_Float16*)alloc((size_t)NB_ * D_ * 2);
  int*      ci      = (int*)     alloc((size_t)NB_ * 4);
  float*    wiv     = (float*)   alloc((size_t)NB_ * 4);
  float*    pu      = (float*)   alloc((size_t)C_ * D_ * 4);
  _Float16* protosT = (_Float16*)alloc((size_t)C_ * D_ * 2);
  float*    z2      = (float*)   alloc((size_t)B_ * C_ * 4);
  int*      topidx  = (int*)     alloc((size_t)B_ * TOPK_ * 4);
  float*    mvec    = (float*)   alloc((size_t)C_ * 4);
  float*    pmacc   = (float*)   alloc((size_t)C_ * 4);
  float*    acc     = (float*)   alloc((size_t)16 * 4);
  _Float16* featsN  = supN + (size_t)N_ * D_;  // l2n(feats) rows live inside supN

  k_zero<<<1, 64, 0, stream>>>(acc, pmacc);
  k_entropy<<<B_, 256, 0, stream>>>(x, ev, acc);
  k_transpose_wf<<<dim3(L_ / 32, D_ / 32), dim3(32, 8), 0, stream>>>(Wf, WfT);
  k_cvt_f16<<<(C_ * D_ + 255) / 256, 256, 0, stream>>>(Wc, Wch, C_ * D_);
  // feats = x @ Wf     (32 rows x 64 cols per wave)
  k_gemm_f32A<<<dim3(D_ / 64, B_ / 128), 128, 0, stream>>>(x, L_, WfT, L_, feats, D_, L_, nullptr);
  // logits = feats @ Wc.T + bc
  k_gemm_f32A<<<dim3(1, B_ / 128), 128, 0, stream>>>(feats, D_, Wch, D_, logits, C_, D_, bc);
  k_softmax<<<B_, 64, 0, stream>>>(logits, probs, logp, pred, pmaxv, acc);
  k_wc<<<B_ / 256, 256, 0, stream>>>(ev, pmaxv, wcv, acc);
  k_m<<<C_, 256, 0, stream>>>(probs, wcv, acc, mvec);
  k_loss1<<<B_, 64, 0, stream>>>(probs, mvec, pmacc, acc);
  k_supnorm<<<NB_, 256, 0, stream>>>(sup_old, feats, lab_old, pred, wcv, supN, ci, wiv);
  k_protos<<<dim3(D_ / 256, C_), 256, 0, stream>>>(supN, ci, wiv, pu);
  k_protos_norm<<<C_, 256, 0, stream>>>(pu, protosT);
  // z2 = l2n(feats) @ protos
  k_gemm_f16A<<<dim3(1, B_ / 64), 128, 0, stream>>>(featsN, D_, protosT, D_, z2, C_, D_);
  k_loss2<<<B_, 64, 0, stream>>>(z2, pred, wcv, acc);
  // fused sim + top-5 (A panel in LDS, dynamic 79,360 B — legal under CDNA5's 320KB/WGP)
  k_sim_topk<<<B_ / 32, 128, SIM_LDS_BYTES, stream>>>(featsN, supN, wcv, topidx);
  k_ncl<<<B_, 64, 0, stream>>>(topidx, sco_old, probs, logp, wcv, acc);
  k_final<<<1, 64, 0, stream>>>(acc, pmacc, out);
}
